// SVDBlock_88819923681717
// MI455X (gfx1250) — matrix operand order repaired and checked by reading, not compiled, not run on hardware
//
#include <hip/hip_runtime.h>
#include <cstdint>
#include <cstddef>

// ---------------------------------------------------------------------------
// MI455X (gfx1250) implementation of the SVD-attention transformer block.
// All matmuls run on v_wmma_f32_16x16x32_bf16 (fp32 accumulate).
// GEMM global->LDS traffic uses the Tensor Data Mover (tensor_load_to_lds)
// with double buffering synchronized via s_wait_tensorcnt.
// ---------------------------------------------------------------------------

typedef __bf16 bf16_t;
typedef __bf16 v8bf  __attribute__((ext_vector_type(8)));
typedef __bf16 v16bf __attribute__((ext_vector_type(16)));
typedef float  v8f   __attribute__((ext_vector_type(8)));
typedef unsigned int u32x4 __attribute__((ext_vector_type(4)));
typedef int    i32x8 __attribute__((ext_vector_type(8)));
typedef int    i32x4 __attribute__((ext_vector_type(4)));

#define DEV static __device__ __forceinline__

static constexpr int BB   = 4;
static constexpr int TT   = 4096;
static constexpr int DD   = 1024;
static constexpr int RR   = 128;
static constexpr int DFF  = 4096;
static constexpr int WIN  = 1024;
static constexpr int MROW = BB * TT;          // 16384

// ---------------------------------------------------------------------------
// WMMA fragment helpers (contiguous k order k = 16h+s, same for A and B, wave32)
// ---------------------------------------------------------------------------
DEV v16bf concat8(v8bf lo, v8bf hi) {
  return __builtin_shufflevector(lo, hi, 0,1,2,3,4,5,6,7,8,9,10,11,12,13,14,15);
}
// A-matrix 16x32 bf16: lane group h=0 holds K=0..15, h=1 holds K=16..31
DEV v16bf load_afrag(const bf16_t* rowPtr, int h) {
  v8bf lo = *(const v8bf*)(rowPtr + h * 16);
  v8bf hi = *(const v8bf*)(rowPtr + h * 16 + 8);
  return concat8(lo, hi);
}
// B-matrix 32x16 bf16: lane n holds column n; h=0 -> K=0..15, h=1 -> K=16..31
DEV v16bf load_bfrag(const bf16_t* rowPtr, int h) {
  v8bf lo = *(const v8bf*)(rowPtr + h * 16);
  v8bf hi = *(const v8bf*)(rowPtr + h * 16 + 8);
  return concat8(lo, hi);
}
DEV v8f wmma_bf16(v16bf a, v16bf b, v8f c) {
  return __builtin_amdgcn_wmma_f32_16x16x32_bf16(false, a, false, b, (short)0, c,
                                                 false, false);
}
DEV float red_max16(float v) {  // reduce across the 16-lane half (N dim of D layout)
  v = fmaxf(v, __shfl_xor(v, 1, 32));
  v = fmaxf(v, __shfl_xor(v, 2, 32));
  v = fmaxf(v, __shfl_xor(v, 4, 32));
  v = fmaxf(v, __shfl_xor(v, 8, 32));
  return v;
}
DEV float red_sum16(float v) {
  v += __shfl_xor(v, 1, 32);
  v += __shfl_xor(v, 2, 32);
  v += __shfl_xor(v, 4, 32);
  v += __shfl_xor(v, 8, 32);
  return v;
}

// ---------------------------------------------------------------------------
// Tensor Data Mover: load a 128x32 bf16 tile (row stride ldK elements) into LDS
// with 16B padding after every 64B row => padded LDS row stride of 40 bf16.
// D# built per CDNA5 ISA 8.3 (group0) / 8.4 (group1). This toolchain's builtin
// is the 6-arg form: (g0, g1, g2, g3, g_extra, cpol) — unused groups zeroed.
// ---------------------------------------------------------------------------
DEV void tdm_load_tile(const bf16_t* base, long long row0, int k0, int ldK,
                       unsigned ldsByteAddr) {
  unsigned long long ga = (unsigned long long)(uintptr_t)base +
                          ((unsigned long long)row0 * (unsigned)ldK + (unsigned)k0) * 2ull;
  u32x4 g0;
  g0[0] = 1u;                                              // count=1 (valid D#)
  g0[1] = ldsByteAddr;                                     // lds_addr (bytes)
  g0[2] = (unsigned)ga;                                    // global_addr[31:0]
  g0[3] = (unsigned)((ga >> 32) & 0x1FFFFFFull) | (2u << 30);  // addr[56:32], type=2
  i32x8 g1;
  // data_size=1 (2B), pad_enable=1, pad_interval=3 (16 DW = 64B), pad_amount=3 (4 DW = 16B)
  g1[0] = (1 << 16) | (1 << 20) | (3 << 22) | (3 << 25);
  g1[1] = 0;                    // tensor_dim0 = 1<<20 -> lo16 = 0 (never OOB for our tiles)
  g1[2] = 16;                   // tensor_dim0 hi16 = 16 ; tensor_dim1 lo16 = 0
  g1[3] = 16 | (32 << 16);      // tensor_dim1 hi16 = 16 ; tile_dim0 = 32
  g1[4] = 128;                  // tile_dim1 = 128 ; tile_dim2 = 0 (2D tile)
  g1[5] = ldK;                  // tensor_dim0_stride lo32 (elements)
  g1[6] = 0;
  g1[7] = 0;
  i32x4 z4; z4[0] = z4[1] = z4[2] = z4[3] = 0;
  i32x8 z8; z8[0] = z8[1] = z8[2] = z8[3] = z8[4] = z8[5] = z8[6] = z8[7] = 0;
  __builtin_amdgcn_tensor_load_to_lds(g0, g1, z4, z4, z8, 0);
}

// ---------------------------------------------------------------------------
// fp32 -> bf16 convert (weight packing)
// ---------------------------------------------------------------------------
__global__ void cvt_bf16_kernel(const float* __restrict__ src,
                                bf16_t* __restrict__ dst, int n) {
  int i = blockIdx.x * blockDim.x + threadIdx.x;
  if (i < n) dst[i] = (bf16_t)src[i];
}

// ---------------------------------------------------------------------------
// RMSNorm over D=1024 -> bf16 output. One row per block, 256 threads.
// ---------------------------------------------------------------------------
__global__ __launch_bounds__(256)
void rmsnorm_bf16_kernel(const float* __restrict__ x, const float* __restrict__ w,
                         bf16_t* __restrict__ out) {
  const long long row = blockIdx.x;
  const float4 v = ((const float4*)(x + row * DD))[threadIdx.x];
  float ss = v.x * v.x + v.y * v.y + v.z * v.z + v.w * v.w;
  ss += __shfl_xor(ss, 1, 32);  ss += __shfl_xor(ss, 2, 32);
  ss += __shfl_xor(ss, 4, 32);  ss += __shfl_xor(ss, 8, 32);
  ss += __shfl_xor(ss, 16, 32);
  __shared__ float red[8];
  if ((threadIdx.x & 31) == 0) red[threadIdx.x >> 5] = ss;
  __syncthreads();
  float tot = red[0] + red[1] + red[2] + red[3] + red[4] + red[5] + red[6] + red[7];
  const float r = rsqrtf(tot * (1.0f / (float)DD) + 1e-6f);
  const float4 wv = ((const float4*)w)[threadIdx.x];
  const long long o = row * DD + (long long)threadIdx.x * 4;
  out[o + 0] = (bf16_t)(v.x * r * wv.x);
  out[o + 1] = (bf16_t)(v.y * r * wv.y);
  out[o + 2] = (bf16_t)(v.z * r * wv.z);
  out[o + 3] = (bf16_t)(v.w * r * wv.w);
}

// ---------------------------------------------------------------------------
// RoPE + pack. qkv_raw: (M,384) fp32 [q|k|v]. Writes q,k as (M,128) bf16
// (q pre-scaled by 1/sqrt(R)), v transposed to (B,R,T) bf16 so the attention
// PV B-fragment reads are contiguous along keys.
// ---------------------------------------------------------------------------
__global__ __launch_bounds__(128)
void rope_pack_kernel(const float* __restrict__ qkv, const float* __restrict__ cosT,
                      const float* __restrict__ sinT, bf16_t* __restrict__ qo,
                      bf16_t* __restrict__ ko, bf16_t* __restrict__ vTo) {
  const int bt = blockIdx.x;
  const int b = bt >> 12;          // T = 4096
  const int t = bt & (TT - 1);
  const float* row = qkv + (long long)bt * 384;
  const int i = threadIdx.x;
  if (i < 64) {
    const float c = cosT[t * 64 + i], s = sinT[t * 64 + i];
    const float SC = 0.0883883476483184f;  // 1/sqrt(128), folded into q
    const float q1 = row[i], q2 = row[i + 64];
    qo[(long long)bt * RR + i]      = (bf16_t)((q1 * c + q2 * s) * SC);
    qo[(long long)bt * RR + i + 64] = (bf16_t)((q2 * c - q1 * s) * SC);
    const float k1 = row[128 + i], k2 = row[192 + i];
    ko[(long long)bt * RR + i]      = (bf16_t)(k1 * c + k2 * s);
    ko[(long long)bt * RR + i + 64] = (bf16_t)(k2 * c - k1 * s);
  }
  vTo[((long long)b * RR + i) * TT + t] = (bf16_t)row[256 + i];
}

// ---------------------------------------------------------------------------
// Generic tiled WMMA GEMM: C(MxN) = A(MxK,bf16) @ W(NxK,bf16)^T
//   MODE 0: outF = A@W^T                     (QKV projection)
//   MODE 1: outF = resid + A@W^T             (Wo / w2 projections)
//   MODE 2: outB = (A@W0^T) * silu(A@W1^T)   (fused w1/w3 + SwiGLU gate)
// Block tile 128x128 (8 waves, each 32x64), BK=32, TDM double-buffered LDS.
// ---------------------------------------------------------------------------
template <int MODE>
__global__ __launch_bounds__(256)
void gemm_wmma(const bf16_t* __restrict__ A, const bf16_t* __restrict__ W0,
               const bf16_t* __restrict__ W1w, const float* __restrict__ resid,
               float* __restrict__ outF, bf16_t* __restrict__ outB,
               int M, int N, int K) {
  (void)M; (void)W1w; (void)resid; (void)outF; (void)outB;
  constexpr int LDT = 40;                       // padded LDS row stride (bf16)
  constexpr int NT  = (MODE == 2) ? 3 : 2;      // TDM tiles per stage
  __shared__ __attribute__((aligned(128))) bf16_t sA[2][128 * LDT];
  __shared__ __attribute__((aligned(128))) bf16_t sW[2][128 * LDT];
  __shared__ __attribute__((aligned(128))) bf16_t sW2[(MODE == 2) ? 2 : 1]
                                                     [(MODE == 2) ? 128 * LDT : 1];
  const int tid  = threadIdx.x;
  const int wave = tid >> 5, lane = tid & 31;
  const int h = lane >> 4, ln = lane & 15;
  const int wm = wave & 3, wn = wave >> 2;      // 4x2 wave grid
  const long long row0 = (long long)blockIdx.x * 128;
  const int col0 = blockIdx.y * 128;
  const int nK = K / 32;

  auto issue = [&](int buf, int kt) {
    tdm_load_tile(A,  row0, kt * 32, K, (unsigned)(size_t)(const void*)&sA[buf][0]);
    tdm_load_tile(W0, col0, kt * 32, K, (unsigned)(size_t)(const void*)&sW[buf][0]);
    if constexpr (MODE == 2)
      tdm_load_tile(W1w, col0, kt * 32, K, (unsigned)(size_t)(const void*)&sW2[buf][0]);
  };

  const v8f zf = {0.f, 0.f, 0.f, 0.f, 0.f, 0.f, 0.f, 0.f};
  v8f acc[2][4], acc2[2][4];
#pragma unroll
  for (int i = 0; i < 2; ++i)
#pragma unroll
    for (int j = 0; j < 4; ++j) { acc[i][j] = zf; acc2[i][j] = zf; }

  if (tid < 32) issue(0, 0);                    // prologue (wave 0, TDM ignores EXEC)
  for (int kt = 0; kt < nK; ++kt) {
    const int buf = kt & 1;
    if (tid < 32) {
      if (kt + 1 < nK) {                        // prefetch next stage, then wait for
        issue(buf ^ 1, kt + 1);                 // current stage (TENSORcnt in-order)
        __builtin_amdgcn_s_wait_tensorcnt(NT);
      } else {
        __builtin_amdgcn_s_wait_tensorcnt(0);
      }
    }
    __syncthreads();
    const bf16_t* aT = sA[buf];
    const bf16_t* wT = sW[buf];
    v16bf af0 = load_afrag(aT + (wm * 32 + 0  + ln) * LDT, h);
    v16bf af1 = load_afrag(aT + (wm * 32 + 16 + ln) * LDT, h);
#pragma unroll
    for (int tn = 0; tn < 4; ++tn) {
      v16bf bfv = load_bfrag(wT + (wn * 64 + tn * 16 + ln) * LDT, h);
      acc[0][tn] = wmma_bf16(af0, bfv, acc[0][tn]);
      acc[1][tn] = wmma_bf16(af1, bfv, acc[1][tn]);
      if constexpr (MODE == 2) {
        v16bf bf2 = load_bfrag(sW2[buf] + (wn * 64 + tn * 16 + ln) * LDT, h);
        acc2[0][tn] = wmma_bf16(af0, bf2, acc2[0][tn]);
        acc2[1][tn] = wmma_bf16(af1, bf2, acc2[1][tn]);
      }
    }
    __syncthreads();                            // everyone done reading buf before
  }                                             // TDM refills it next iteration

  // Epilogue. D layout: elem r -> row r+8h, col = lane&15 of each 16x16 subtile.
#pragma unroll
  for (int tm = 0; tm < 2; ++tm)
#pragma unroll
    for (int tn = 0; tn < 4; ++tn)
#pragma unroll
      for (int r = 0; r < 8; ++r) {
        const long long gr = row0 + wm * 32 + tm * 16 + r + 8 * h;
        const int gc = col0 + wn * 64 + tn * 16 + ln;
        const long long idx = gr * (long long)N + gc;
        const float v = acc[tm][tn][r];
        if constexpr (MODE == 0) outF[idx] = v;
        if constexpr (MODE == 1) outF[idx] = resid[idx] + v;
        if constexpr (MODE == 2) {
          const float s = acc2[tm][tn][r];
          outB[idx] = (bf16_t)(v * (s / (1.0f + __expf(-s))));
        }
      }
}

// ---------------------------------------------------------------------------
// Windowed-causal flash attention. One wave owns 16 query rows; iterates
// 32-key tiles in [row-WIN, row]. S = QK^T via 8 WMMAs/tile, online softmax
// with shfl_xor row reductions (stats align with the D layout), probs staged
// through per-wave LDS into an A fragment, PV via 8 WMMAs/tile.
// ---------------------------------------------------------------------------
__global__ __launch_bounds__(256)
void attn_wmma(const bf16_t* __restrict__ q, const bf16_t* __restrict__ k,
               const bf16_t* __restrict__ vT, bf16_t* __restrict__ y) {
  __shared__ __attribute__((aligned(128))) bf16_t pst[8][16 * 32];
  const int tid = threadIdx.x, wave = tid >> 5, lane = tid & 31;
  const int h = lane >> 4, ln = lane & 15;
  const int tile = blockIdx.x * 8 + wave;
  const int tilesPerB = TT / 16;
  const int b  = tile / tilesPerB;
  const int qb = (tile % tilesPerB) * 16;
  const bf16_t* qB = q  + (long long)b * TT * RR;
  const bf16_t* kB = k  + (long long)b * TT * RR;
  const bf16_t* vB = vT + (long long)b * RR * TT;

  v16bf qf[4];
  {
    const bf16_t* qr = qB + (long long)(qb + ln) * RR;
#pragma unroll
    for (int c = 0; c < 4; ++c) qf[c] = load_afrag(qr + c * 32, h);
  }

  const v8f zf = {0.f, 0.f, 0.f, 0.f, 0.f, 0.f, 0.f, 0.f};
  v8f acc[8];
  float mrow[8], lrow[8];
#pragma unroll
  for (int j = 0; j < 8; ++j) acc[j] = zf;
#pragma unroll
  for (int r = 0; r < 8; ++r) { mrow[r] = -1e30f; lrow[r] = 0.f; }

  int s_begin = qb - WIN; if (s_begin < 0) s_begin = 0;
  s_begin &= ~31;
  for (int s0 = s_begin; s0 <= qb + 15; s0 += 32) {
    v8f S0 = zf, S1 = zf;
#pragma unroll
    for (int c = 0; c < 4; ++c) {   // K chunks of 32 over R=128
      v16bf b0 = load_bfrag(kB + (long long)(s0 + ln) * RR + c * 32, h);
      v16bf b1 = load_bfrag(kB + (long long)(s0 + 16 + ln) * RR + c * 32, h);
      S0 = wmma_bf16(qf[c], b0, S0);
      S1 = wmma_bf16(qf[c], b1, S1);
    }
    float al[8];
#pragma unroll
    for (int r = 0; r < 8; ++r) {
      const int row = qb + r + 8 * h;
      const int c0 = s0 + ln, c1 = c0 + 16;
      float s0v = ((c0 <= row) && (row - c0 <= WIN)) ? S0[r] : -1e30f;
      float s1v = ((c1 <= row) && (row - c1 <= WIN)) ? S1[r] : -1e30f;
      const float tmax = red_max16(fmaxf(s0v, s1v));
      const float mn = fmaxf(mrow[r], tmax);
      const float a = __expf(mrow[r] - mn);     // alpha=0 wipes any sentinel garbage
      mrow[r] = mn; al[r] = a;
      s0v = __expf(s0v - mn);
      s1v = __expf(s1v - mn);
      lrow[r] = lrow[r] * a + red_sum16(s0v + s1v);
      pst[wave][(r + 8 * h) * 32 + ln]      = (bf16_t)s0v;   // stage P (16x32)
      pst[wave][(r + 8 * h) * 32 + 16 + ln] = (bf16_t)s1v;
    }
#pragma unroll
    for (int j = 0; j < 8; ++j)
#pragma unroll
      for (int r = 0; r < 8; ++r) acc[j][r] *= al[r];
    const v16bf pa = load_afrag(&pst[wave][ln * 32], h);     // per-wave, DScnt-ordered
#pragma unroll
    for (int j = 0; j < 8; ++j) {                            // out cols j*16..j*16+15
      const bf16_t* vr = vB + (long long)(j * 16 + ln) * TT + s0 + h * 16;
      acc[j] = wmma_bf16(pa, concat8(*(const v8bf*)vr, *(const v8bf*)(vr + 8)), acc[j]);
    }
  }
#pragma unroll
  for (int r = 0; r < 8; ++r) {
    const float inv = 1.0f / lrow[r];
    const long long o = ((long long)b * TT + qb + r + 8 * h) * RR;
#pragma unroll
    for (int j = 0; j < 8; ++j) y[o + j * 16 + ln] = (bf16_t)(acc[j][r] * inv);
  }
}

// ---------------------------------------------------------------------------
// Host orchestration
// ---------------------------------------------------------------------------
extern "C" void kernel_launch(void* const* d_in, const int* in_sizes, int n_in,
                              void* d_out, int out_size, void* d_ws, size_t ws_size,
                              hipStream_t stream) {
  (void)in_sizes; (void)n_in; (void)out_size; (void)ws_size;
  const float* x    = (const float*)d_in[0];
  const float* Uq   = (const float*)d_in[1];
  const float* Uk   = (const float*)d_in[2];
  const float* Uv   = (const float*)d_in[3];
  const float* cosT = (const float*)d_in[4];
  const float* sinT = (const float*)d_in[5];
  const float* ln1  = (const float*)d_in[6];
  const float* ln2  = (const float*)d_in[7];
  const float* Wo   = (const float*)d_in[8];
  const float* w1   = (const float*)d_in[9];
  const float* w3   = (const float*)d_in[10];
  const float* w2   = (const float*)d_in[11];
  float* out = (float*)d_out;

  char* p = (char*)d_ws;
  auto carve = [&](size_t bytes) -> void* {
    void* r = (void*)p;
    p += (bytes + 255) & ~(size_t)255;
    return r;
  };
  bf16_t* wU   = (bf16_t*)carve((size_t)3 * RR * DD * 2);     // [Uq;Uk;Uv] packed
  bf16_t* wWo  = (bf16_t*)carve((size_t)DD * RR * 2);
  bf16_t* wW1  = (bf16_t*)carve((size_t)DFF * DD * 2);
  bf16_t* wW3  = (bf16_t*)carve((size_t)DFF * DD * 2);
  bf16_t* wW2  = (bf16_t*)carve((size_t)DD * DFF * 2);
  bf16_t* nrm  = (bf16_t*)carve((size_t)MROW * DD * 2);       // reused for h
  float*  qkvR = (float*) carve((size_t)MROW * 3 * RR * 4);
  bf16_t* qb   = (bf16_t*)carve((size_t)MROW * RR * 2);
  bf16_t* kb   = (bf16_t*)carve((size_t)MROW * RR * 2);
  bf16_t* vTb  = (bf16_t*)carve((size_t)BB * RR * TT * 2);
  bf16_t* yb   = (bf16_t*)carve((size_t)MROW * RR * 2);
  float*  x1   = (float*) carve((size_t)MROW * DD * 4);
  bf16_t* u    = (bf16_t*)carve((size_t)MROW * DFF * 2);

  auto cvt = [&](const float* s, bf16_t* d, int n) {
    cvt_bf16_kernel<<<(n + 255) / 256, 256, 0, stream>>>(s, d, n);
  };
  cvt(Uq, wU + 0 * RR * DD, RR * DD);
  cvt(Uk, wU + 1 * RR * DD, RR * DD);
  cvt(Uv, wU + 2 * RR * DD, RR * DD);
  cvt(Wo, wWo, DD * RR);
  cvt(w1, wW1, DFF * DD);
  cvt(w3, wW3, DFF * DD);
  cvt(w2, wW2, DD * DFF);

  // 1) normed = rmsnorm(x, ln1) in bf16
  rmsnorm_bf16_kernel<<<MROW, 256, 0, stream>>>(x, ln1, nrm);
  // 2) qkv_raw = normed @ [Uq;Uk;Uv]^T   (M=16384, N=384, K=1024)
  gemm_wmma<0><<<dim3(MROW / 128, 384 / 128), 256, 0, stream>>>(
      nrm, wU, nullptr, nullptr, qkvR, nullptr, MROW, 384, DD);
  // 3) RoPE + pack (q scaled, v transposed)
  rope_pack_kernel<<<MROW, 128, 0, stream>>>(qkvR, cosT, sinT, qb, kb, vTb);
  // 4) windowed flash attention -> y (bf16)
  attn_wmma<<<(MROW / 16) / 8, 256, 0, stream>>>(qb, kb, vTb, yb);
  // 5) x1 = x + y @ Wo^T   (N=1024, K=128)
  gemm_wmma<1><<<dim3(MROW / 128, DD / 128), 256, 0, stream>>>(
      yb, wWo, nullptr, x, x1, nullptr, MROW, DD, RR);
  // 6) h = rmsnorm(x1, ln2) in bf16 (reuse nrm)
  rmsnorm_bf16_kernel<<<MROW, 256, 0, stream>>>(x1, ln2, nrm);
  // 7) u = (h @ w1^T) * silu(h @ w3^T)   (N=4096, K=1024)
  gemm_wmma<2><<<dim3(MROW / 128, DFF / 128), 256, 0, stream>>>(
      nrm, wW1, wW3, nullptr, nullptr, u, MROW, DFF, DD);
  // 8) out = x1 + u @ w2^T   (N=1024, K=4096)
  gemm_wmma<1><<<dim3(MROW / 128, DD / 128), 256, 0, stream>>>(
      u, wW2, nullptr, x1, out, nullptr, MROW, DD, DFF);
}
